// CaptioningTransformer_21191368639301
// MI455X (gfx1250) — matrix operand order repaired and checked by reading, not compiled, not run on hardware
//
#include <hip/hip_runtime.h>
#include <cstdint>
#include <cstddef>

// ---------------- problem dims (fixed by the reference) ----------------
#define N_  32
#define T_  256
#define S_  512
#define D_  512
#define H_  8
#define F_  200
#define L_  4
#define V_  10000
#define HD_ 64            // D / H

// ---------------- WMMA types ----------------
typedef __attribute__((ext_vector_type(16))) __bf16 v16bf;
typedef __attribute__((ext_vector_type(8)))  float  v8f;

union FragB {
  unsigned short u[16];
  uint4          q[2];
  v16bf          v;
};

struct Tile { FragB a0, a1, b0, b1; };

__device__ __forceinline__ unsigned short f32_to_bf16(float f) {
  unsigned int u = __float_as_uint(f);
  u += 0x7FFFu + ((u >> 16) & 1u);          // round-to-nearest-even
  return (unsigned short)(u >> 16);
}

// Guarded fragment loaders (cold paths: tile edges / K tails only).
__device__ __forceinline__ void loadA_g(FragB& f, const unsigned short* __restrict__ base,
                                        bool ok, int ka, int K) {
  if (ok && (ka + 16 <= K)) {
    f.q[0] = *(const uint4*)(base + ka);
    f.q[1] = *(const uint4*)(base + ka + 8);
  } else {
#pragma unroll
    for (int j = 0; j < 8; ++j) {
      int k = ka + j, k2 = ka + 8 + j;
      f.u[j]     = (ok && k  < K) ? base[k ] : (unsigned short)0;
      f.u[8 + j] = (ok && k2 < K) ? base[k2] : (unsigned short)0;
    }
  }
}
__device__ __forceinline__ void loadB_g(FragB& f, const unsigned short* __restrict__ base,
                                        bool ok, int kb, int K) {
  if (ok && (kb + 16 <= K)) {
    f.q[0] = *(const uint4*)(base + kb);
    f.q[1] = *(const uint4*)(base + kb + 8);
  } else {
#pragma unroll
    for (int j = 0; j < 16; ++j) {
      int k = kb + j;
      f.u[j] = (ok && k < K) ? base[k] : (unsigned short)0;
    }
  }
}

// Hot-path tile loader: 8 guard-free b128 loads.
__device__ __forceinline__ void load_tile(Tile& t,
    const unsigned short* __restrict__ Ab, const unsigned short* __restrict__ Bb,
    long long aoff0, long long aoff1, long long boff0, long long boff1,
    int k0, int lhi)
{
  const int ka = k0 + 16 * lhi;
  const int kb = k0 + 16 * lhi;
  t.a0.q[0] = *(const uint4*)(Ab + aoff0 + ka);
  t.a0.q[1] = *(const uint4*)(Ab + aoff0 + ka + 8);
  t.a1.q[0] = *(const uint4*)(Ab + aoff1 + ka);
  t.a1.q[1] = *(const uint4*)(Ab + aoff1 + ka + 8);
  t.b0.q[0] = *(const uint4*)(Bb + boff0 + kb);
  t.b0.q[1] = *(const uint4*)(Bb + boff0 + kb + 8);
  t.b1.q[0] = *(const uint4*)(Bb + boff1 + kb);
  t.b1.q[1] = *(const uint4*)(Bb + boff1 + kb + 8);
}

__device__ __forceinline__ void mma_tile(v8f acc[2][2], const Tile& t)
{
  acc[0][0] = __builtin_amdgcn_wmma_f32_16x16x32_bf16(false, t.a0.v, false, t.b0.v, (short)0, acc[0][0], false, false);
  acc[0][1] = __builtin_amdgcn_wmma_f32_16x16x32_bf16(false, t.a0.v, false, t.b1.v, (short)0, acc[0][1], false, false);
  acc[1][0] = __builtin_amdgcn_wmma_f32_16x16x32_bf16(false, t.a1.v, false, t.b0.v, (short)0, acc[1][0], false, false);
  acc[1][1] = __builtin_amdgcn_wmma_f32_16x16x32_bf16(false, t.a1.v, false, t.b1.v, (short)0, acc[1][1], false, false);
}

// =====================================================================
// Batched GEMM:  C = alpha * (A bf16 [MxK] @ B bf16 [KxN]) + bias
//   * A row-major bf16;  B supplied TRANSPOSED (Bt, N-major, row len ldb>=K)
//   * outMode: 0 = f32 store, 1 = bf16 store,
//              2 = bf16 store transposed into (b, h, d, s) with ldc = Tk
//                  (fuses the attention V-head transpose into the epilogue)
//   * two-level batch z = zb*bH + zh with independent strides
// Block: 256 thr = 8 waves; block tile 128x64; wave tile 32x32 = 4 WMMAs
// per 32-K step from 8 b128 loads.  Interior blocks run a guard-free,
// SOFTWARE-PIPELINED (double-buffered) K-loop so next-tile loads overlap
// the current tile's WMMAs; edge blocks take a cold guarded loop.
// Fragment k order: contiguous, k = 16*(lane/16) + s, same for A and B.
// =====================================================================
__global__ __launch_bounds__(256) void gemm_bf16_wmma(
    const unsigned short* __restrict__ A, int lda, long long sAb, long long sAh,
    const unsigned short* __restrict__ Bt, int ldb, long long sBb, long long sBh,
    void* __restrict__ Cv, int ldc, long long sCb, long long sCh,
    int M, int N, int K,
    const float* __restrict__ bias, float alpha, int relu, int outMode, int bH)
{
  const int z  = blockIdx.z;
  const int zb = z / bH, zh = z % bH;
  const unsigned short* Ab = A  + zb * sAb + zh * sAh;
  const unsigned short* Bb = Bt + zb * sBb + zh * sBh;

  const int wave = threadIdx.x >> 5;
  const int lane = threadIdx.x & 31;
  const int l16  = lane & 15;
  const int lhi  = lane >> 4;

  const int rowTile = blockIdx.y * 128;
  const int colTile = blockIdx.x * 64;
  const int rowBase = rowTile + (wave & 3) * 32;   // wave covers 32 rows
  const int colBase = colTile + (wave >> 2) * 32;  // wave covers 32 cols

  const long long aoff0 = (long long)(rowBase + l16)      * lda;
  const long long aoff1 = (long long)(rowBase + 16 + l16) * lda;
  const long long boff0 = (long long)(colBase + l16)      * ldb;
  const long long boff1 = (long long)(colBase + 16 + l16) * ldb;

  v8f acc[2][2] = {{{}, {}}, {{}, {}}};

  const bool interior = (rowTile + 128 <= M) && (colTile + 64 <= N);
  const int  Kfull    = K & ~31;

  if (interior) {
    // ---- hot path: guard-free, double-buffered K-loop ----
    const int nk = Kfull >> 5;          // # of full 32-K tiles
    if (nk > 0) {
      Tile t0, t1;
      load_tile(t0, Ab, Bb, aoff0, aoff1, boff0, boff1, 0, lhi);
      int i = 1;
      for (; i + 1 < nk; i += 2) {
        load_tile(t1, Ab, Bb, aoff0, aoff1, boff0, boff1, i * 32, lhi);
        mma_tile(acc, t0);
        if ((i + 3) * 32 < K) {         // stream-ahead hint (global_prefetch)
          __builtin_prefetch((const void*)(Ab + aoff0 + (i + 3) * 32), 0, 1);
          __builtin_prefetch((const void*)(Bb + boff0 + (i + 3) * 32), 0, 1);
        }
        load_tile(t0, Ab, Bb, aoff0, aoff1, boff0, boff1, (i + 1) * 32, lhi);
        mma_tile(acc, t1);
      }
      if (i < nk) {                      // one remaining full tile
        load_tile(t1, Ab, Bb, aoff0, aoff1, boff0, boff1, i * 32, lhi);
        mma_tile(acc, t0);
        mma_tile(acc, t1);
      } else {
        mma_tile(acc, t0);
      }
    }
    if (Kfull < K) {                     // K tail (e.g. K=200): k-guards only
      const int ka = Kfull + 16 * lhi;
      const int kb = Kfull + 16 * lhi;
      Tile t;
      loadA_g(t.a0, Ab + aoff0, true, ka, K);
      loadA_g(t.a1, Ab + aoff1, true, ka, K);
      loadB_g(t.b0, Bb + boff0, true, kb, K);
      loadB_g(t.b1, Bb + boff1, true, kb, K);
      mma_tile(acc, t);
    }
  } else {
    // ---- cold path: fully guarded (M/N fringe blocks) ----
    const bool r0 = (rowBase + l16)      < M;
    const bool r1 = (rowBase + 16 + l16) < M;
    const bool c0 = (colBase + l16)      < N;
    const bool c1 = (colBase + 16 + l16) < N;
    for (int k0 = 0; k0 < K; k0 += 32) {
      const int ka = k0 + 16 * lhi;
      const int kb = k0 + 16 * lhi;
      Tile t;
      loadA_g(t.a0, Ab + aoff0, r0, ka, K);
      loadA_g(t.a1, Ab + aoff1, r1, ka, K);
      loadB_g(t.b0, Bb + boff0, c0, kb, K);
      loadB_g(t.b1, Bb + boff1, c1, kb, K);
      mma_tile(acc, t);
    }
  }

  // ---- epilogue: scale, bias, relu, store (f32 / bf16 / bf16-transposed) ----
  float*          Cf = (float*)Cv          + zb * sCb + zh * sCh;
  unsigned short* Ch = (unsigned short*)Cv + zb * sCb + zh * sCh;
  const int tkShift = (outMode == 2) ? (31 - __clz(ldc)) : 0;  // ldc=Tk pow2
#pragma unroll
  for (int mi = 0; mi < 2; ++mi) {
#pragma unroll
    for (int ni = 0; ni < 2; ++ni) {
      const int col = colBase + 16 * ni + l16;
      float bv = (bias != nullptr && col < N) ? bias[col] : 0.f;
#pragma unroll
      for (int i = 0; i < 8; ++i) {
        const int r = rowBase + 16 * mi + 8 * lhi + i;
        if (interior || (r < M && col < N)) {
          float v = acc[mi][ni][i] * alpha + bv;
          if (relu) v = fmaxf(v, 0.f);
          if (outMode == 0) {
            Cf[(long long)r * ldc + col] = v;
          } else if (outMode == 1) {
            Ch[(long long)r * ldc + col] = f32_to_bf16(v);
          } else {
            // row r = (b, s), col = h*64 + d  ->  Vt[b][h][d][s], ldc = Tk
            const int b = r >> tkShift, s = r & (ldc - 1);
            const int h = col >> 6,    d = col & 63;
            const long long idx = (((long long)b * H_ + h) * HD_ + d) * ldc + s;
            Ch[idx] = f32_to_bf16(v);
          }
        }
      }
    }
  }
}

// ------------- weight f32 (R x C, batched) -> bf16 transposed (C x R) -------------
__global__ __launch_bounds__(256) void wt_cvt_kernel(const float* __restrict__ W,
    unsigned short* __restrict__ Wt, int R, int Cc, long long total)
{
  long long i = (long long)blockIdx.x * blockDim.x + threadIdx.x;
  if (i >= total) return;
  int c       = (int)(i % Cc);
  long long t = i / Cc;
  int r       = (int)(t % R);
  long long l = t / R;
  Wt[l * (long long)R * Cc + (long long)c * R + r] = f32_to_bf16(W[i]);
}

// ------------- x = emb[captions] + PE(t,d); writes f32 residual + bf16 mirror -------------
__global__ __launch_bounds__(256) void embed_pe_kernel(const float* __restrict__ emb,
    const int* __restrict__ captions, float* __restrict__ x, unsigned short* __restrict__ xb)
{
  long long i = (long long)blockIdx.x * blockDim.x + threadIdx.x;
  const long long total = (long long)N_ * T_ * D_;
  if (i >= total) return;
  int d        = (int)(i % D_);
  long long nt = i / D_;
  int t        = (int)(nt % T_);
  int i2   = d & ~1;
  float dv = expf((float)i2 * (-9.210340371976184f / (float)D_));   // 10000^(-i2/D)
  float an = (float)t * dv;
  float pe = (d & 1) ? cosf(an) : sinf(an);
  int tok  = captions[nt];
  float v  = emb[(long long)tok * D_ + d] + pe;
  x[i]  = v;
  xb[i] = f32_to_bf16(v);
}

// ------------- memory = features + PE(s,d), bf16 (only used as GEMM A) -------------
__global__ __launch_bounds__(256) void add_pe_kernel(const float* __restrict__ feat,
    unsigned short* __restrict__ mem)
{
  long long i = (long long)blockIdx.x * blockDim.x + threadIdx.x;
  const long long total = (long long)N_ * S_ * D_;
  if (i >= total) return;
  int d = (int)(i % D_);
  int s = (int)((i / D_) % S_);
  int i2   = d & ~1;
  float dv = expf((float)i2 * (-9.210340371976184f / (float)D_));
  float an = (float)s * dv;
  float pe = (d & 1) ? cosf(an) : sinf(an);
  mem[i] = f32_to_bf16(feat[i] + pe);
}

// ------------- row softmax (optional causal & key-pad mask), bf16 probs out -------------
// scores layout: (z = b*H + h, t, 0..Tk-1); grid = (T_, N_*H_)
__global__ __launch_bounds__(256) void softmax_mask_kernel(const float* __restrict__ Sc,
    unsigned short* __restrict__ P, int Tk, int causal, const int* __restrict__ captions)
{
  __shared__ float red[256];
  const int t = blockIdx.x;
  const int z = blockIdx.y;
  const int b = z / H_;
  const long long rowoff = ((long long)z * T_ + t) * Tk;
  const float* row = Sc + rowoff;
  unsigned short* prow = P + rowoff;
  const int tid = threadIdx.x;

  float vals[2];
  bool  ok[2];
  float m = -INFINITY;
#pragma unroll
  for (int e = 0; e < 2; ++e) {
    int j = tid + e * 256;
    bool v = (j < Tk);
    if (v && causal) v = (j <= t) && (captions[b * T_ + j] != 0);
    ok[e]   = v;
    vals[e] = v ? row[j] : -INFINITY;
    m = fmaxf(m, vals[e]);
  }
  red[tid] = m; __syncthreads();
  for (int s2 = 128; s2 > 0; s2 >>= 1) {
    if (tid < s2) red[tid] = fmaxf(red[tid], red[tid + s2]);
    __syncthreads();
  }
  m = red[0]; __syncthreads();

  float sum = 0.f;
#pragma unroll
  for (int e = 0; e < 2; ++e) {
    vals[e] = ok[e] ? expf(vals[e] - m) : 0.f;
    sum += vals[e];
  }
  red[tid] = sum; __syncthreads();
  for (int s2 = 128; s2 > 0; s2 >>= 1) {
    if (tid < s2) red[tid] += red[tid + s2];
    __syncthreads();
  }
  float inv = 1.f / red[0];
#pragma unroll
  for (int e = 0; e < 2; ++e) {
    int j = tid + e * 256;
    if (j < Tk) prow[j] = f32_to_bf16(vals[e] * inv);
  }
}

// ------------- x = LayerNorm(x + delta)*g + b; writes f32 residual + bf16 mirror -------------
__global__ __launch_bounds__(256) void ln_residual_kernel(float* __restrict__ x,
    const float* __restrict__ delta, const float* __restrict__ g,
    const float* __restrict__ beta, unsigned short* __restrict__ xb)
{
  __shared__ float red[256];
  const long long row = blockIdx.x;
  float* xr = x + row * D_;
  const float* dr = delta + row * D_;
  unsigned short* xbr = xb + row * D_;
  const int tid = threadIdx.x;

  float v0 = xr[tid] + dr[tid];
  float v1 = xr[tid + 256] + dr[tid + 256];
  red[tid] = v0 + v1; __syncthreads();
  for (int s2 = 128; s2 > 0; s2 >>= 1) {
    if (tid < s2) red[tid] += red[tid + s2];
    __syncthreads();
  }
  float mu = red[0] * (1.f / 512.f); __syncthreads();
  float d0 = v0 - mu, d1 = v1 - mu;
  red[tid] = d0 * d0 + d1 * d1; __syncthreads();
  for (int s2 = 128; s2 > 0; s2 >>= 1) {
    if (tid < s2) red[tid] += red[tid + s2];
    __syncthreads();
  }
  float rs = rsqrtf(red[0] * (1.f / 512.f) + 1e-5f);
  float o0 = d0 * rs * g[tid]       + beta[tid];
  float o1 = d1 * rs * g[tid + 256] + beta[tid + 256];
  xr[tid]        = o0;
  xr[tid + 256]  = o1;
  xbr[tid]       = f32_to_bf16(o0);
  xbr[tid + 256] = f32_to_bf16(o1);
}

// =====================================================================
// Host orchestration
// =====================================================================
extern "C" void kernel_launch(void* const* d_in, const int* in_sizes, int n_in,
                              void* d_out, int out_size, void* d_ws, size_t ws_size,
                              hipStream_t stream)
{
  (void)in_sizes; (void)n_in; (void)out_size; (void)ws_size;
  // setup_inputs() dict order, params flattened in insertion order:
  const float* features = (const float*)d_in[0];
  const float* emb      = (const float*)d_in[1];
  const float* sWq = (const float*)d_in[2];
  const float* sWk = (const float*)d_in[3];
  const float* sWv = (const float*)d_in[4];
  const float* sWo = (const float*)d_in[5];
  const float* sbq = (const float*)d_in[6];
  const float* sbk = (const float*)d_in[7];
  const float* sbv = (const float*)d_in[8];
  const float* sbo = (const float*)d_in[9];
  const float* cWq = (const float*)d_in[10];
  const float* cWk = (const float*)d_in[11];
  const float* cWv = (const float*)d_in[12];
  const float* cWo = (const float*)d_in[13];
  const float* cbq = (const float*)d_in[14];
  const float* cbk = (const float*)d_in[15];
  const float* cbv = (const float*)d_in[16];
  const float* cbo = (const float*)d_in[17];
  const float* W1  = (const float*)d_in[18];
  const float* b1  = (const float*)d_in[19];
  const float* W2  = (const float*)d_in[20];
  const float* b2  = (const float*)d_in[21];
  const float* ln1g = (const float*)d_in[22];
  const float* ln1b = (const float*)d_in[23];
  const float* ln2g = (const float*)d_in[24];
  const float* ln2b = (const float*)d_in[25];
  const float* ln3g = (const float*)d_in[26];
  const float* ln3b = (const float*)d_in[27];
  const float* outW = (const float*)d_in[28];
  const float* outb = (const float*)d_in[29];
  const int* captions = (const int*)d_in[30];
  float* out = (float*)d_out;

  // ---- carve workspace ----
  char* ws = (char*)d_ws;
  size_t off = 0;
  auto carve = [&](size_t bytes) -> void* {
    void* p = ws + off;
    off += (bytes + 255) & ~(size_t)255;
    return p;
  };
  unsigned short* wsqt  = (unsigned short*)carve(2ull * L_ * D_ * D_);
  unsigned short* wskt  = (unsigned short*)carve(2ull * L_ * D_ * D_);
  unsigned short* wsvt  = (unsigned short*)carve(2ull * L_ * D_ * D_);
  unsigned short* wsot  = (unsigned short*)carve(2ull * L_ * D_ * D_);
  unsigned short* wcqt  = (unsigned short*)carve(2ull * L_ * D_ * D_);
  unsigned short* wckt  = (unsigned short*)carve(2ull * L_ * D_ * D_);
  unsigned short* wcvtw = (unsigned short*)carve(2ull * L_ * D_ * D_);
  unsigned short* wcot  = (unsigned short*)carve(2ull * L_ * D_ * D_);
  unsigned short* w1t   = (unsigned short*)carve(2ull * L_ * F_ * D_);
  unsigned short* w2t   = (unsigned short*)carve(2ull * L_ * D_ * F_);
  unsigned short* wot   = (unsigned short*)carve(2ull * V_ * D_);
  float*          xbuf  = (float*)carve(4ull * N_ * T_ * D_);
  unsigned short* xbf   = (unsigned short*)carve(2ull * N_ * T_ * D_);
  unsigned short* membf = (unsigned short*)carve(2ull * N_ * S_ * D_);
  unsigned short* qbf   = (unsigned short*)carve(2ull * N_ * T_ * D_);
  unsigned short* kbf   = (unsigned short*)carve(2ull * N_ * S_ * D_);
  unsigned short* vtbf  = (unsigned short*)carve(2ull * N_ * S_ * D_);
  float*          scores= (float*)carve(4ull * N_ * H_ * T_ * S_);
  unsigned short* pbf   = (unsigned short*)carve(2ull * N_ * H_ * T_ * S_);
  unsigned short* ybf   = (unsigned short*)carve(2ull * N_ * T_ * D_);
  float*          attbuf= (float*)carve(4ull * N_ * T_ * D_);
  unsigned short* hbf   = (unsigned short*)carve(2ull * N_ * T_ * F_);

  auto nblk = [](long long n) { return (unsigned)((n + 255) / 256); };
  auto gemm = [&](const unsigned short* A, int lda, long long sAb, long long sAh,
                  const unsigned short* Bt, int ldb, long long sBb, long long sBh,
                  void* C, int ldc, long long sCb, long long sCh,
                  int M, int Ncols, int K, const float* bias,
                  float alpha, int relu, int outMode, int batch, int bH) {
    dim3 grid((Ncols + 63) / 64, (M + 127) / 128, batch);
    gemm_bf16_wmma<<<grid, dim3(256), 0, stream>>>(A, lda, sAb, sAh,
        Bt, ldb, sBb, sBh, C, ldc, sCb, sCh, M, Ncols, K, bias, alpha, relu, outMode, bH);
  };

  // ---- convert all weights to transposed bf16 (per call; deterministic) ----
  const long long wDD = (long long)L_ * D_ * D_;
  wt_cvt_kernel<<<nblk(wDD), 256, 0, stream>>>(sWq, wsqt, D_, D_, wDD);
  wt_cvt_kernel<<<nblk(wDD), 256, 0, stream>>>(sWk, wskt, D_, D_, wDD);
  wt_cvt_kernel<<<nblk(wDD), 256, 0, stream>>>(sWv, wsvt, D_, D_, wDD);
  wt_cvt_kernel<<<nblk(wDD), 256, 0, stream>>>(sWo, wsot, D_, D_, wDD);
  wt_cvt_kernel<<<nblk(wDD), 256, 0, stream>>>(cWq, wcqt, D_, D_, wDD);
  wt_cvt_kernel<<<nblk(wDD), 256, 0, stream>>>(cWk, wckt, D_, D_, wDD);
  wt_cvt_kernel<<<nblk(wDD), 256, 0, stream>>>(cWv, wcvtw, D_, D_, wDD);
  wt_cvt_kernel<<<nblk(wDD), 256, 0, stream>>>(cWo, wcot, D_, D_, wDD);
  const long long wDF = (long long)L_ * D_ * F_;
  wt_cvt_kernel<<<nblk(wDF), 256, 0, stream>>>(W1, w1t, D_, F_, wDF);   // -> (F x D)
  wt_cvt_kernel<<<nblk(wDF), 256, 0, stream>>>(W2, w2t, F_, D_, wDF);   // -> (D x F)
  const long long wDV = (long long)D_ * V_;
  wt_cvt_kernel<<<nblk(wDV), 256, 0, stream>>>(outW, wot, D_, V_, wDV); // -> (V x D)

  // ---- embeddings + positional encodings ----
  const long long nTD = (long long)N_ * T_ * D_;
  const long long nSD = (long long)N_ * S_ * D_;
  embed_pe_kernel<<<nblk(nTD), 256, 0, stream>>>(emb, captions, xbuf, xbf);
  add_pe_kernel<<<nblk(nSD), 256, 0, stream>>>(features, membf);

  for (int l = 0; l < L_; ++l) {
    const long long wo = (long long)l * D_ * D_;
    // -------- self attention --------
    gemm(xbf, D_, 0, 0, wsqt + wo, D_, 0, 0, qbf, D_, 0, 0, N_*T_, D_, D_, sbq + l*D_, 1.f, 0, 1, 1, 1);
    gemm(xbf, D_, 0, 0, wskt + wo, D_, 0, 0, kbf, D_, 0, 0, N_*T_, D_, D_, sbk + l*D_, 1.f, 0, 1, 1, 1);
    // V projection with fused head-transpose: writes Vt (b, h, d, s), ldc = T_
    gemm(xbf, D_, 0, 0, wsvt + wo, D_, 0, 0, vtbf, T_, 0, 0, N_*T_, D_, D_, sbv + l*D_, 1.f, 0, 2, 1, 1);
    // scores = (Q Kt)/sqrt(64); batched over z = b*H + h
    gemm(qbf, D_, (long long)T_*D_, HD_,
         kbf, D_, (long long)T_*D_, HD_,
         scores, T_, (long long)H_*T_*T_, (long long)T_*T_,
         T_, T_, HD_, nullptr, 0.125f, 0, 0, N_*H_, H_);
    softmax_mask_kernel<<<dim3(T_, N_*H_), 256, 0, stream>>>(scores, pbf, T_, 1, captions);
    // y = P V (bf16 out, head-interleaved into (N*T, D))
    gemm(pbf, T_, (long long)H_*T_*T_, (long long)T_*T_,
         vtbf, T_, (long long)H_*HD_*T_, (long long)HD_*T_,
         ybf, D_, (long long)T_*D_, HD_,
         T_, HD_, T_, nullptr, 1.f, 0, 1, N_*H_, H_);
    gemm(ybf, D_, 0, 0, wsot + wo, D_, 0, 0, attbuf, D_, 0, 0, N_*T_, D_, D_, sbo + l*D_, 1.f, 0, 0, 1, 1);
    ln_residual_kernel<<<N_*T_, 256, 0, stream>>>(xbuf, attbuf, ln1g + l*D_, ln1b + l*D_, xbf);

    // -------- cross attention --------
    gemm(xbf,   D_, 0, 0, wcqt  + wo, D_, 0, 0, qbf, D_, 0, 0, N_*T_, D_, D_, cbq + l*D_, 1.f, 0, 1, 1, 1);
    gemm(membf, D_, 0, 0, wckt  + wo, D_, 0, 0, kbf, D_, 0, 0, N_*S_, D_, D_, cbk + l*D_, 1.f, 0, 1, 1, 1);
    gemm(membf, D_, 0, 0, wcvtw + wo, D_, 0, 0, vtbf, S_, 0, 0, N_*S_, D_, D_, cbv + l*D_, 1.f, 0, 2, 1, 1);
    gemm(qbf, D_, (long long)T_*D_, HD_,
         kbf, D_, (long long)S_*D_, HD_,
         scores, S_, (long long)H_*T_*S_, (long long)T_*S_,
         T_, S_, HD_, nullptr, 0.125f, 0, 0, N_*H_, H_);
    softmax_mask_kernel<<<dim3(T_, N_*H_), 256, 0, stream>>>(scores, pbf, S_, 0, captions);
    gemm(pbf, S_, (long long)H_*T_*S_, (long long)T_*S_,
         vtbf, S_, (long long)H_*HD_*S_, (long long)HD_*S_,
         ybf, D_, (long long)T_*D_, HD_,
         T_, HD_, S_, nullptr, 1.f, 0, 1, N_*H_, H_);
    gemm(ybf, D_, 0, 0, wcot + wo, D_, 0, 0, attbuf, D_, 0, 0, N_*T_, D_, D_, cbo + l*D_, 1.f, 0, 0, 1, 1);
    ln_residual_kernel<<<N_*T_, 256, 0, stream>>>(xbuf, attbuf, ln2g + l*D_, ln2b + l*D_, xbf);

    // -------- feed-forward --------
    gemm(xbf, D_, 0, 0, w1t + (long long)l*F_*D_, D_, 0, 0, hbf, F_, 0, 0,
         N_*T_, F_, D_, b1 + l*F_, 1.f, 1, 1, 1, 1);
    gemm(hbf, F_, 0, 0, w2t + (long long)l*D_*F_, F_, 0, 0, attbuf, D_, 0, 0,
         N_*T_, D_, F_, b2 + l*D_, 1.f, 0, 0, 1, 1);
    ln_residual_kernel<<<N_*T_, 256, 0, stream>>>(xbuf, attbuf, ln3g + l*D_, ln3b + l*D_, xbf);
  }

  // -------- output head: logits = x @ outW + outb (f32 to d_out) --------
  gemm(xbf, D_, 0, 0, wot, D_, 0, 0, out, V_, 0, 0, N_*T_, V_, D_, outb, 1.f, 0, 0, 1, 1);
}
